// NERLSTM_CRF_12017318494629
// MI455X (gfx1250) — matrix operand order repaired and checked by reading, not compiled, not run on hardware
//
#include <hip/hip_runtime.h>
#include <math.h>

// ---------------- problem constants ----------------
#define BB   64      // batch
#define TT   256     // seq len
#define EE   256     // embed dim
#define HD   512     // per-direction hidden
#define G4   2048    // 4*HD gate width
#define HH   1024    // 2*HD
#define KK   33      // tags
#define KP   48      // padded tags (3 WMMA tiles)
#define BT   16384   // B*T rows (time-major: row = t*64 + b)

typedef __attribute__((ext_vector_type(16))) _Float16 v16h;
typedef __attribute__((ext_vector_type(8)))  _Float16 v8h;
typedef __attribute__((ext_vector_type(8)))  float    v8f;

union V16U { v16h v; v8h h[2]; };

// A-fragment (16x32 f16): lane m = lane&15; chunks at koff and koff+16,
// koff = k0 + 8*(lane>>4). `p` is row base (row stride folded in by caller).
__device__ __forceinline__ v16h frag_a(const _Float16* __restrict__ p, int koff) {
  V16U r;
  r.h[0] = *(const v8h*)(p + koff);
  r.h[1] = *(const v8h*)(p + koff + 16);
  return r.v;
}
// B-fragment (32x16 f16): lane n = lane&15; chunks at koff and koff+16,
// koff = k0 + 8*(lane>>4). `p` is column (= weight-row) base.
__device__ __forceinline__ v16h frag_b(const _Float16* __restrict__ p, int koff) {
  V16U r;
  r.h[0] = *(const v8h*)(p + koff);
  r.h[1] = *(const v8h*)(p + koff + 16);
  return r.v;
}

__device__ __forceinline__ v8f wmma32(v16h a, v16h b, v8f c) {
  return __builtin_amdgcn_wmma_f32_16x16x32_f16(
      /*neg_a=*/false, a, /*neg_b=*/false, b,
      /*c_mod=*/(short)0, c, /*reuse_a=*/false, /*reuse_b=*/false);
}

__device__ __forceinline__ float sigm(float x) { return 1.0f / (1.0f + __expf(-x)); }

// ---------------- prep kernels ----------------
__global__ void f32_to_f16_kernel(const float* __restrict__ s, _Float16* __restrict__ d, int n) {
  int i = blockIdx.x * blockDim.x + threadIdx.x;
  if (i < n) d[i] = (_Float16)s[i];
}

__global__ void pad_wout_kernel(const float* __restrict__ w, _Float16* __restrict__ d) {
  int i = blockIdx.x * blockDim.x + threadIdx.x;   // KP*HH
  if (i >= KP * HH) return;
  int r = i >> 10, k = i & 1023;
  d[i] = (r < KK) ? (_Float16)w[r * HH + k] : (_Float16)0.0f;
}

__global__ void bias_kernel(const float* __restrict__ bihf, const float* __restrict__ bhhf,
                            const float* __restrict__ bihb, const float* __restrict__ bhhb,
                            const float* __restrict__ bout,
                            float* __restrict__ bf, float* __restrict__ bb,
                            float* __restrict__ bop) {
  int i = blockIdx.x * blockDim.x + threadIdx.x;
  if (i < G4) { bf[i] = bihf[i] + bhhf[i]; bb[i] = bihb[i] + bhhb[i]; }
  if (i < KP) bop[i] = (i < KK) ? bout[i] : 0.0f;
}

__global__ void init_state_kernel(_Float16* __restrict__ h0,
                                  float* __restrict__ cf, float* __restrict__ cb) {
  int i = blockIdx.x * blockDim.x + threadIdx.x;   // BB*HD
  if (i < BB * HD) { h0[i] = (_Float16)0.0f; cf[i] = 0.0f; cb[i] = 0.0f; }
}

// Embedding gather, time-major rows: xe[(t*64+b)*256 + e], f32 -> f16.
__global__ void embed_gather_kernel(const int* __restrict__ x,
                                    const float* __restrict__ embed,
                                    _Float16* __restrict__ xe) {
  int id = blockIdx.x * blockDim.x + threadIdx.x;  // BT * 64 threads, 4 elems each
  int row = id >> 6;
  int e4  = (id & 63) << 2;
  int t = row >> 6, b = row & 63;
  int tok = x[b * TT + t];
  float4 v = *(const float4*)(embed + (size_t)tok * EE + e4);
  _Float16* o = xe + (size_t)row * EE + e4;
  o[0] = (_Float16)v.x; o[1] = (_Float16)v.y; o[2] = (_Float16)v.z; o[3] = (_Float16)v.w;
}

// ---------------- input GEMM: gx = xe @ w_ih^T + bias  (f16 out) ----------------
// grid (BT/16, G4/(16*8), 2); block 256 (8 waves, one 16x16 N-tile each)
__global__ __launch_bounds__(256) void input_gemm_kernel(
    const _Float16* __restrict__ xe,
    const _Float16* __restrict__ wih0, const _Float16* __restrict__ wih1,
    const float* __restrict__ bias0, const float* __restrict__ bias1,
    _Float16* __restrict__ gx0, _Float16* __restrict__ gx1) {
  int dir = blockIdx.z;
  const _Float16* wih  = dir ? wih1  : wih0;
  const float*    bias = dir ? bias1 : bias0;
  _Float16*       gx   = dir ? gx1   : gx0;

  int tid = threadIdx.x, lane = tid & 31, wave = tid >> 5;
  int m0 = blockIdx.x * 16;
  int n0 = (blockIdx.y * 8 + wave) * 16;
  int l16 = lane & 15, hi = lane >> 4;
  const _Float16* ap = xe  + (size_t)(m0 + l16) * EE;
  const _Float16* bp = wih + (size_t)(n0 + l16) * EE;
  int aoff = hi << 3, boff = hi << 3;

  v8f acc = {};
#pragma unroll
  for (int kk = 0; kk < EE / 32; ++kk) {
    v16h a = frag_a(ap, kk * 32 + aoff);
    v16h b = frag_b(bp, kk * 32 + boff);
    acc = wmma32(a, b, acc);
  }
  int col = n0 + l16;
  float bv = bias[col];
  int rowb = m0 + (hi << 3);
#pragma unroll
  for (int r = 0; r < 8; ++r)
    gx[(size_t)(rowb + r) * G4 + col] = (_Float16)(acc[r] + bv);
}

// ---------------- fused LSTM step (both directions per launch) ----------------
// grid (HD/16, 2); block 512 (16 waves). wave = (mt<<2)|gate: each wave does the
// 16x16 tile (rows b = mt*16.., cols = gate*512 + cg*16..) of h_prev @ w_hh^T,
// seeded from gx; gates combined via LDS into c/h update.
__global__ __launch_bounds__(512) void lstm_step_kernel(
    const _Float16* __restrict__ gx0, const _Float16* __restrict__ gx1,
    const _Float16* __restrict__ whh0, const _Float16* __restrict__ whh1,
    _Float16* __restrict__ hs0, _Float16* __restrict__ hs1,
    const _Float16* __restrict__ h0,
    float* __restrict__ c0, float* __restrict__ c1, int s) {
  __shared__ float gl[4 * BB * 16];   // [gate][b][nl]

  int dir = blockIdx.y;
  int t = dir ? (TT - 1 - s) : s;
  const _Float16* gx  = dir ? gx1  : gx0;
  const _Float16* whh = dir ? whh1 : whh0;
  _Float16* hs = dir ? hs1 : hs0;
  float*    c  = dir ? c1  : c0;
  const _Float16* hprev;
  if (s == 0) hprev = h0;
  else        hprev = dir ? (hs1 + (size_t)(t + 1) * BB * HD)
                          : (hs0 + (size_t)(t - 1) * BB * HD);

  int tid = threadIdx.x, lane = tid & 31, wave = tid >> 5;
  int mt = wave >> 2, gate = wave & 3, cg = blockIdx.x;
  int n0 = gate * HD + cg * 16;
  int m0 = mt * 16;
  int l16 = lane & 15, hi = lane >> 4;

  const _Float16* ap = hprev + (size_t)(m0 + l16) * HD;
  const _Float16* bp = whh   + (size_t)(n0 + l16) * HD;
  int aoff = hi << 3, boff = hi << 3;
  __builtin_prefetch(ap, 0, 0);   // global_prefetch_b8

  // seed accumulator with hoisted input-GEMM gates
  v8f acc;
  {
    int col  = n0 + l16;
    int rowb = t * BB + m0 + (hi << 3);
#pragma unroll
    for (int r = 0; r < 8; ++r)
      acc[r] = (float)gx[(size_t)(rowb + r) * G4 + col];
  }
#pragma unroll
  for (int kk = 0; kk < HD / 32; ++kk) {
    int k0 = kk * 32;
    v16h a = frag_a(ap, k0 + aoff);
    v16h b = frag_b(bp, k0 + boff);
    acc = wmma32(a, b, acc);
  }
  {
    int mb = m0 + (hi << 3);
#pragma unroll
    for (int r = 0; r < 8; ++r)
      gl[(gate * BB + mb + r) * 16 + l16] = acc[r];
  }
  __syncthreads();

  // elementwise i/f/g/o -> c,h : 64 rows x 16 cols = 1024 values, 2 per thread
  int e = tid;
#pragma unroll
  for (int q = 0; q < 2; ++q, e += 512) {
    int b_ = e >> 4, nl = e & 15;
    int col = cg * 16 + nl;
    float gi = gl[(0 * BB + b_) * 16 + nl];
    float gf = gl[(1 * BB + b_) * 16 + nl];
    float gg = gl[(2 * BB + b_) * 16 + nl];
    float go = gl[(3 * BB + b_) * 16 + nl];
    float cold = c[b_ * HD + col];
    float cn = sigm(gf) * cold + sigm(gi) * tanhf(gg);
    float h  = sigm(go) * tanhf(cn);
    c[b_ * HD + col] = cn;
    hs[(size_t)t * BB * HD + b_ * HD + col] = (_Float16)h;
  }
}

// ---------------- emission GEMM: em = [hf|hb] @ w_out^T + b_out, masked ----------------
// grid (BT/16); block 96 (3 waves, one padded N-tile each)
__global__ __launch_bounds__(96) void emission_kernel(
    const _Float16* __restrict__ hsf, const _Float16* __restrict__ hsb,
    const _Float16* __restrict__ woutp, const float* __restrict__ boutp,
    const float* __restrict__ mask, float* __restrict__ em) {
  int tid = threadIdx.x, lane = tid & 31, wave = tid >> 5;
  int m0 = blockIdx.x * 16;
  int n0 = wave * 16;
  int l16 = lane & 15, hi = lane >> 4;
  int arow = m0 + l16;
  const _Float16* bp = woutp + (size_t)(n0 + l16) * HH;
  int aoff = hi << 3, boff = hi << 3;

  v8f acc = {};
#pragma unroll 4
  for (int kk = 0; kk < HH / 32; ++kk) {
    int k0 = kk * 32;
    int ka = k0 + aoff;
    int k2 = ka + 16;
    const _Float16* s1 = (ka < HD) ? (hsf + (size_t)arow * HD + ka)
                                   : (hsb + (size_t)arow * HD + (ka - HD));
    const _Float16* s2 = (k2 < HD) ? (hsf + (size_t)arow * HD + k2)
                                   : (hsb + (size_t)arow * HD + (k2 - HD));
    V16U a; a.h[0] = *(const v8h*)s1; a.h[1] = *(const v8h*)s2;
    v16h b = frag_b(bp, k0 + boff);
    acc = wmma32(a.v, b, acc);
  }
  int col = n0 + l16;
  float bv = boutp[col];
  int rowb = m0 + (hi << 3);
#pragma unroll
  for (int r = 0; r < 8; ++r) {
    int row = rowb + r;
    int bb = row & 63, t = row >> 6;
    float mv = mask[bb * TT + t];
    em[(size_t)row * KP + col] = (acc[r] + bv) * mv;
  }
}

// ---------------- CRF Viterbi: one workgroup per batch element ----------------
__global__ __launch_bounds__(64) void viterbi_kernel(
    const float* __restrict__ em, const float* __restrict__ start_t,
    const float* __restrict__ end_t, const float* __restrict__ trans,
    const float* __restrict__ mask, int* __restrict__ hist,
    float* __restrict__ out) {
  __shared__ float sc[KP];
  __shared__ float tl[KK * KK];
  int b = blockIdx.x, j = threadIdx.x;

  for (int i = j; i < KK * KK; i += 64) tl[i] = trans[i];
  if (j < KK) sc[j] = start_t[j] + em[(size_t)(0 * BB + b) * KP + j];
  __syncthreads();

  for (int t = 1; t < TT; ++t) {
    float best = -1e30f; int bi = 0;
    bool m = mask[b * TT + t] > 0.5f;
    if (j < KK) {
      for (int i = 0; i < KK; ++i) {
        float v = sc[i] + tl[i * KK + j];
        if (v > best) { best = v; bi = i; }
      }
      hist[(size_t)((t - 1) * BB + b) * KP + j] = bi;
    }
    __syncthreads();
    if (j < KK && m) sc[j] = best + em[(size_t)(t * BB + b) * KP + j];
    __syncthreads();
  }

  if (j == 0) {
    float best = -1e30f; int last = 0;
    for (int jj = 0; jj < KK; ++jj) {
      float v = sc[jj] + end_t[jj];
      if (v > best) { best = v; last = jj; }
    }
    out[b * TT + (TT - 1)] = (float)last;
    int tag = last;
    for (int t = TT - 2; t >= 0; --t) {
      bool m = mask[b * TT + (t + 1)] > 0.5f;
      if (m) tag = hist[(size_t)(t * BB + b) * KP + tag];
      out[b * TT + t] = (float)tag;
    }
  }
}

// ---------------- host ----------------
extern "C" void kernel_launch(void* const* d_in, const int* in_sizes, int n_in,
                              void* d_out, int out_size, void* d_ws, size_t ws_size,
                              hipStream_t stream) {
  const int*   x       = (const int*)  d_in[0];
  const float* mask    = (const float*)d_in[1];
  const float* embed   = (const float*)d_in[2];
  const float* w_ih_f  = (const float*)d_in[3];
  const float* w_hh_f  = (const float*)d_in[4];
  const float* b_ih_f  = (const float*)d_in[5];
  const float* b_hh_f  = (const float*)d_in[6];
  const float* w_ih_b  = (const float*)d_in[7];
  const float* w_hh_b  = (const float*)d_in[8];
  const float* b_ih_b  = (const float*)d_in[9];
  const float* b_hh_b  = (const float*)d_in[10];
  const float* w_out   = (const float*)d_in[11];
  const float* b_out   = (const float*)d_in[12];
  const float* start_t = (const float*)d_in[13];
  const float* end_t   = (const float*)d_in[14];
  const float* trans   = (const float*)d_in[15];
  float* out = (float*)d_out;

  char* base = (char*)d_ws; size_t off = 0;
  auto take = [&](size_t bytes) -> char* {
    char* p = base + off; off = (off + bytes + 255) & ~(size_t)255; return p;
  };
  _Float16* xe    = (_Float16*)take((size_t)BT * EE * 2);
  _Float16* wihf  = (_Float16*)take((size_t)G4 * EE * 2);
  _Float16* wihb  = (_Float16*)take((size_t)G4 * EE * 2);
  _Float16* whhf  = (_Float16*)take((size_t)G4 * HD * 2);
  _Float16* whhb  = (_Float16*)take((size_t)G4 * HD * 2);
  _Float16* woutp = (_Float16*)take((size_t)KP * HH * 2);
  float*    biasf = (float*)   take((size_t)G4 * 4);
  float*    biasb = (float*)   take((size_t)G4 * 4);
  float*    boutp = (float*)   take((size_t)KP * 4);
  _Float16* gxf   = (_Float16*)take((size_t)BT * G4 * 2);
  _Float16* gxb   = (_Float16*)take((size_t)BT * G4 * 2);
  _Float16* hsf   = (_Float16*)take((size_t)BT * HD * 2);
  _Float16* hsb   = (_Float16*)take((size_t)BT * HD * 2);
  _Float16* h0    = (_Float16*)take((size_t)BB * HD * 2);
  float*    cf    = (float*)   take((size_t)BB * HD * 4);
  float*    cb    = (float*)   take((size_t)BB * HD * 4);
  float*    em    = (float*)   take((size_t)BT * KP * 4);
  int*      hist  = (int*)     take((size_t)TT * BB * KP * 4);

  // prep: weight conversions, bias fusion, state init
  f32_to_f16_kernel<<<(G4 * EE + 255) / 256, 256, 0, stream>>>(w_ih_f, wihf, G4 * EE);
  f32_to_f16_kernel<<<(G4 * EE + 255) / 256, 256, 0, stream>>>(w_ih_b, wihb, G4 * EE);
  f32_to_f16_kernel<<<(G4 * HD + 255) / 256, 256, 0, stream>>>(w_hh_f, whhf, G4 * HD);
  f32_to_f16_kernel<<<(G4 * HD + 255) / 256, 256, 0, stream>>>(w_hh_b, whhb, G4 * HD);
  pad_wout_kernel<<<(KP * HH + 255) / 256, 256, 0, stream>>>(w_out, woutp);
  bias_kernel<<<(G4 + 255) / 256, 256, 0, stream>>>(b_ih_f, b_hh_f, b_ih_b, b_hh_b,
                                                    b_out, biasf, biasb, boutp);
  init_state_kernel<<<(BB * HD + 255) / 256, 256, 0, stream>>>(h0, cf, cb);

  // embedding gather (time-major)
  embed_gather_kernel<<<(BT * 64) / 256, 256, 0, stream>>>(x, embed, xe);

  // hoisted input GEMMs, both directions
  input_gemm_kernel<<<dim3(BT / 16, G4 / (16 * 8), 2), 256, 0, stream>>>(
      xe, wihf, wihb, biasf, biasb, gxf, gxb);

  // sequential recurrent steps (fwd t=s, bwd t=255-s fused per launch)
  for (int s = 0; s < TT; ++s)
    lstm_step_kernel<<<dim3(HD / 16, 2), 512, 0, stream>>>(
        gxf, gxb, whhf, whhb, hsf, hsb, h0, cf, cb, s);

  // emissions + CRF decode
  emission_kernel<<<BT / 16, 96, 0, stream>>>(hsf, hsb, woutp, boutp, mask, em);
  viterbi_kernel<<<BB, 64, 0, stream>>>(em, start_t, end_t, trans, mask, hist, out);
}
